// Decoder_66812511257046
// MI455X (gfx1250) — compile-verified, matrix operand order checked by reading, not run on hardware
//
#include <hip/hip_runtime.h>

// ---------------------------------------------------------------------------
// ProbSparse attention decoder for MI455X (gfx1250).
// 8 GEMMs (16384x512)@(512x512) dominate -> bf16 WMMA, LDS double buffering,
// async global->LDS staging via GLOBAL_LOAD_ASYNC_TO_LDS_B128 (ASYNCcnt).
// ---------------------------------------------------------------------------

typedef __bf16 bf16_t;
typedef __attribute__((ext_vector_type(16))) __bf16 v16bf;
typedef __attribute__((ext_vector_type(8)))  __bf16 v8bf;
typedef __attribute__((ext_vector_type(8)))  float  v8f;
typedef int v4i __attribute__((vector_size(16)));

#define DM    512
#define LSEQ  4096
#define BATCH 4
#define NH    8
#define DKH   64
#define NSAMP 27

#define BM 128
#define BN 128
#define BK 32
#define LDA 40   // bf16 pitch: 80 B rows -> 16B-aligned b128, bank-skewed
#define LDB 40

#if __has_builtin(__builtin_amdgcn_global_load_async_to_lds_b128) && \
    __has_builtin(__builtin_amdgcn_s_wait_asynccnt)
#define HAVE_ASYNC 1
#endif

__device__ __forceinline__ bf16_t f2bf(float f) {
  unsigned u = __builtin_bit_cast(unsigned, f);
  unsigned r = u + 0x7FFFu + ((u >> 16) & 1u);   // round-to-nearest-even
  unsigned short h = (unsigned short)(r >> 16);
  return __builtin_bit_cast(bf16_t, h);
}

#ifdef HAVE_ASYNC
typedef __attribute__((address_space(1))) v4i* gv4i_p;
typedef __attribute__((address_space(3))) v4i* lv4i_p;

__device__ __forceinline__ void async_b128(const bf16_t* g, bf16_t* l) {
  __builtin_amdgcn_global_load_async_to_lds_b128(
      (gv4i_p)(__attribute__((address_space(1))) void*)const_cast<bf16_t*>(g),
      (lv4i_p)(__attribute__((address_space(3))) void*)l,
      0, 0);
}
#endif

// ---- one K-step of WMMA work on a staged tile pair -------------------------
__device__ __forceinline__ void wmma_tile(const bf16_t* Ab, const bf16_t* Bb,
                                          int wm, int wn, int rsel, int ksel,
                                          v8f acc[2][4]) {
  v16bf af[2], bfrag[4];
#pragma unroll
  for (int mi = 0; mi < 2; ++mi) {
    const bf16_t* p = Ab + (wm * 32 + mi * 16 + rsel) * LDA + ksel;
    v8bf lo = *(const v8bf*)p;
    v8bf hi = *(const v8bf*)(p + 16);
#pragma unroll
    for (int e = 0; e < 8; ++e) { af[mi][e] = lo[e]; af[mi][8 + e] = hi[e]; }
  }
#pragma unroll
  for (int ni = 0; ni < 4; ++ni) {
    const bf16_t* p = Bb + (wn * 64 + ni * 16 + rsel) * LDB + ksel;
    v8bf lo = *(const v8bf*)p;
    v8bf hi = *(const v8bf*)(p + 16);
#pragma unroll
    for (int e = 0; e < 8; ++e) { bfrag[ni][e] = lo[e]; bfrag[ni][8 + e] = hi[e]; }
  }
#pragma unroll
  for (int mi = 0; mi < 2; ++mi)
#pragma unroll
    for (int ni = 0; ni < 4; ++ni)
      acc[mi][ni] = __builtin_amdgcn_wmma_f32_16x16x32_bf16(
          false, af[mi], false, bfrag[ni], (short)0, acc[mi][ni], false, false);
}

// ---------------------------------------------------------------------------
// C[16384 x 512] = A(bf16) @ Wt(bf16, [n][k]) + bias
// optional fp32 out1, bf16 out1bf, and out2[row2] = C + xs[row2] (residual).
// ---------------------------------------------------------------------------
__global__ __launch_bounds__(256)
void gemm_bf16_kernel(const bf16_t* __restrict__ A, const bf16_t* __restrict__ Wt,
                      const float* __restrict__ bias,
                      float* __restrict__ out1, bf16_t* __restrict__ out1bf,
                      float* __restrict__ out2, const float* __restrict__ xs,
                      int row_off) {
  __shared__ __align__(16) bf16_t As[2][BM * LDA];
  __shared__ __align__(16) bf16_t Bs[2][BN * LDB];

  const int tid  = threadIdx.x;
  const int lane = tid & 31;
  const int wid  = tid >> 5;
  const int wm   = wid >> 1;          // 0..3 -> 32-row strips
  const int wn   = wid & 1;           // 0..1 -> 64-col strips
  const int m0   = blockIdx.y * BM;
  const int n0   = blockIdx.x * BN;

  // staging coordinates: thread copies 32B of A row + 32B of Wt row per tile
  const int srow = tid >> 1;                 // 0..127
  const int soff = (tid & 1) * 16;           // element offset 0 / 16
  const bf16_t* Ag = A  + (size_t)(m0 + srow) * DM + soff;
  const bf16_t* Wg = Wt + (size_t)(n0 + srow) * DM + soff;
  bf16_t* Al0 = &As[0][srow * LDA + soff];
  bf16_t* Bl0 = &Bs[0][srow * LDB + soff];
  const int ABUF = BM * LDA, BBUF = BN * LDB;

  const int rsel = lane & 15;
  const int ksel = (lane >> 4) << 3;         // 0 or 8

  v8f acc[2][4];
#pragma unroll
  for (int mi = 0; mi < 2; ++mi)
#pragma unroll
    for (int ni = 0; ni < 4; ++ni)
#pragma unroll
      for (int r = 0; r < 8; ++r) acc[mi][ni][r] = 0.0f;

  constexpr int NT = DM / BK;   // 16 K-tiles

#ifdef HAVE_ASYNC
  // ---- async double-buffered pipeline (ASYNCcnt-tracked DMA to LDS) -------
  async_b128(Ag, Al0); async_b128(Ag + 8, Al0 + 8);
  async_b128(Wg, Bl0); async_b128(Wg + 8, Bl0 + 8);
  for (int it = 0; it < NT; ++it) {
    const int cur = it & 1;
    if (it + 1 < NT) {
      const bf16_t* an = Ag + (it + 1) * BK;
      const bf16_t* wg2 = Wg + (it + 1) * BK;
      bf16_t* al = Al0 + ((it + 1) & 1) * ABUF;
      bf16_t* bl = Bl0 + ((it + 1) & 1) * BBUF;
      async_b128(an, al); async_b128(an + 8, al + 8);
      async_b128(wg2, bl); async_b128(wg2 + 8, bl + 8);
      __builtin_amdgcn_s_wait_asynccnt(4);   // tile `it` landed; next in flight
    } else {
      __builtin_amdgcn_s_wait_asynccnt(0);
    }
    __syncthreads();
    wmma_tile(&As[cur][0], &Bs[cur][0], wm, wn, rsel, ksel, acc);
    __syncthreads();                         // safe to overwrite buf `cur` next
  }
#else
  // ---- synchronous register-pipelined double buffer -----------------------
  uint4 ra0 = *(const uint4*)Ag;
  uint4 ra1 = *(const uint4*)(Ag + 8);
  uint4 rw0 = *(const uint4*)Wg;
  uint4 rw1 = *(const uint4*)(Wg + 8);
  *(uint4*)Al0 = ra0; *(uint4*)(Al0 + 8) = ra1;
  *(uint4*)Bl0 = rw0; *(uint4*)(Bl0 + 8) = rw1;
  __syncthreads();
  for (int it = 0; it < NT; ++it) {
    const int cur = it & 1;
    const bool more = (it + 1 < NT);
    if (more) {
      const bf16_t* an = Ag + (it + 1) * BK;
      const bf16_t* wg2 = Wg + (it + 1) * BK;
      ra0 = *(const uint4*)an;  ra1 = *(const uint4*)(an + 8);
      rw0 = *(const uint4*)wg2; rw1 = *(const uint4*)(wg2 + 8);
    }
    wmma_tile(&As[cur][0], &Bs[cur][0], wm, wn, rsel, ksel, acc);
    if (more) {
      bf16_t* al = Al0 + ((it + 1) & 1) * ABUF;
      bf16_t* bl = Bl0 + ((it + 1) & 1) * BBUF;
      *(uint4*)al = ra0; *(uint4*)(al + 8) = ra1;
      *(uint4*)bl = rw0; *(uint4*)(bl + 8) = rw1;
    }
    __syncthreads();
  }
#endif

  // ---- epilogue: C lane layout (col = lane&15, row = r + 8*(lane>>4)) -----
  const int rbase = (lane >> 4) << 3;
#pragma unroll
  for (int mi = 0; mi < 2; ++mi) {
#pragma unroll
    for (int ni = 0; ni < 4; ++ni) {
      int col = n0 + wn * 64 + ni * 16 + (lane & 15);
      float bcol = bias[col];
#pragma unroll
      for (int r = 0; r < 8; ++r) {
        int row = m0 + wm * 32 + mi * 16 + rbase + r;
        float v = acc[mi][ni][r] + bcol;
        if (out1)   out1[(size_t)row * DM + col] = v;
        if (out1bf) out1bf[(size_t)row * DM + col] = f2bf(v);
        if (out2) {
          int b = row >> 12, l = row & (LSEQ - 1);
          size_t r2 = (size_t)(b * 2 * LSEQ + row_off + l) * DM + col;
          out2[r2] = v + xs[r2];
        }
      }
    }
  }
}

// ---------------------------------------------------------------------------
// fp32 -> bf16 (groups of 8, vector stores)
// ---------------------------------------------------------------------------
__global__ void cvt_bf16_kernel(const float* __restrict__ src,
                                bf16_t* __restrict__ dst) {
  int g = blockIdx.x * 256 + threadIdx.x;     // exact grid, 8 elems/thread
  const float4* s = (const float4*)(src + ((size_t)g << 3));
  float4 a = s[0], b = s[1];
  bf16_t tmp[8] = {f2bf(a.x), f2bf(a.y), f2bf(a.z), f2bf(a.w),
                   f2bf(b.x), f2bf(b.y), f2bf(b.z), f2bf(b.w)};
  *(uint4*)(dst + ((size_t)g << 3)) = *(uint4*)tmp;
}

// ---------------------------------------------------------------------------
// W[k][n] fp32 -> Wt[n][k] bf16 (32x32 LDS transpose tiles)
// ---------------------------------------------------------------------------
__global__ __launch_bounds__(256)
void wt_convert_kernel(const float* __restrict__ W, bf16_t* __restrict__ Wt) {
  __shared__ float t[32][33];
  const int n0 = blockIdx.x * 32, k0 = blockIdx.y * 32;
  const int tx = threadIdx.x & 31, ty = threadIdx.x >> 5;   // 32 x 8
#pragma unroll
  for (int j = 0; j < 32; j += 8)
    t[ty + j][tx] = W[(size_t)(k0 + ty + j) * DM + n0 + tx];
  __syncthreads();
#pragma unroll
  for (int j = 0; j < 32; j += 8)
    Wt[(size_t)(n0 + ty + j) * DM + k0 + tx] = f2bf(t[tx][ty + j]);
}

// ---------------------------------------------------------------------------
// Threefry-2x32: reproduces jax.random.randint(key,(4096,27),0,4096)
// ---------------------------------------------------------------------------
__device__ __forceinline__ unsigned rotl32(unsigned x, int n) {
  return (x << n) | (x >> (32 - n));
}

__global__ void threefry_idx_kernel(int* __restrict__ sidx, unsigned k0,
                                    unsigned k1, int total) {
  int i = blockIdx.x * blockDim.x + threadIdx.x;
  int half = total >> 1;
  if (i >= half) return;
  unsigned ks2 = 0x1BD11BDAu ^ k0 ^ k1;
  unsigned ksA[3] = {k0, k1, ks2};
  unsigned x0 = (unsigned)i + k0;
  unsigned x1 = (unsigned)(i + half) + k1;
  const int r0[4] = {13, 15, 26, 6};
  const int r1[4] = {17, 29, 16, 24};
#pragma unroll
  for (int blk = 0; blk < 5; ++blk) {
    const int* rr = (blk & 1) ? r1 : r0;
#pragma unroll
    for (int j = 0; j < 4; ++j) { x0 += x1; x1 = rotl32(x1, rr[j]); x1 ^= x0; }
    x0 += ksA[(blk + 1) % 3];
    x1 += ksA[(blk + 2) % 3] + (unsigned)(blk + 1);
  }
  sidx[i]        = (int)(x0 & (LSEQ - 1));
  sidx[i + half] = (int)(x1 & (LSEQ - 1));
}

// ---------------------------------------------------------------------------
// M[b,h,l] = max_s q.k_s  -  (sum_s q.k_s)/L_K   over 27 sampled keys
// ---------------------------------------------------------------------------
__global__ __launch_bounds__(256)
void sample_score_kernel(const float* __restrict__ Q, const float* __restrict__ K,
                         const int* __restrict__ sidx, float* __restrict__ Mout) {
  int t = blockIdx.x * 256 + threadIdx.x;
  int l = t & (LSEQ - 1);
  int h = (t >> 12) & (NH - 1);
  int b = t >> 15;
  const float4* q4 = (const float4*)(Q + (size_t)(b * LSEQ + l) * DM + h * DKH);
  float4 qr[16];
#pragma unroll
  for (int i = 0; i < 16; ++i) qr[i] = q4[i];
  float mx = -3.4e38f, sm = 0.f;
  for (int s = 0; s < NSAMP; ++s) {
    int ki = sidx[l * NSAMP + s];
    const float4* k4 = (const float4*)(K + (size_t)(b * LSEQ + ki) * DM + h * DKH);
    float d = 0.f;
#pragma unroll
    for (int i = 0; i < 16; ++i) {
      float4 a = qr[i], c = k4[i];
      d += a.x * c.x + a.y * c.y + a.z * c.z + a.w * c.w;
    }
    mx = fmaxf(mx, d);
    sm += d;
  }
  Mout[t] = mx - sm * (1.0f / (float)LSEQ);
}

// ---------------------------------------------------------------------------
// Iterative top-27 (lowest-index tie-break), per (b,h)
// ---------------------------------------------------------------------------
__global__ __launch_bounds__(256)
void topk_kernel(const float* __restrict__ Mv, int* __restrict__ topidx) {
  __shared__ float sv[LSEQ];
  __shared__ float rv[256];
  __shared__ int   ri[256];
  const int bh = blockIdx.x;
  const float* m = Mv + (size_t)bh * LSEQ;
  for (int i = threadIdx.x; i < LSEQ; i += 256) sv[i] = m[i];
  __syncthreads();
  for (int it = 0; it < NSAMP; ++it) {
    float best = -3.4e38f; int bi = LSEQ;
    for (int i = threadIdx.x; i < LSEQ; i += 256) {
      float v = sv[i];
      if (v > best || (v == best && i < bi)) { best = v; bi = i; }
    }
    rv[threadIdx.x] = best; ri[threadIdx.x] = bi;
    __syncthreads();
    for (int off = 128; off; off >>= 1) {
      if (threadIdx.x < off) {
        float v2 = rv[threadIdx.x + off]; int i2 = ri[threadIdx.x + off];
        if (v2 > rv[threadIdx.x] ||
            (v2 == rv[threadIdx.x] && i2 < ri[threadIdx.x])) {
          rv[threadIdx.x] = v2; ri[threadIdx.x] = i2;
        }
      }
      __syncthreads();
    }
    if (threadIdx.x == 0) { topidx[bh * NSAMP + it] = ri[0]; sv[ri[0]] = -3.4e38f; }
    __syncthreads();
  }
}

// ---------------------------------------------------------------------------
// vmean[b,h,d] = mean_l V[b,l,h,d]
// ---------------------------------------------------------------------------
__global__ __launch_bounds__(256)
void vmean_kernel(const float* __restrict__ V, float* __restrict__ vmean) {
  __shared__ float parts[256];
  const int bh = blockIdx.x;
  const int b = bh >> 3, h = bh & 7;
  const int d = threadIdx.x & 63, c = threadIdx.x >> 6;
  float s = 0.f;
  for (int l = c; l < LSEQ; l += 4)
    s += V[(size_t)(b * LSEQ + l) * DM + h * DKH + d];
  parts[threadIdx.x] = s;
  __syncthreads();
  if (threadIdx.x < 64) {
    float t = parts[threadIdx.x] + parts[64 + threadIdx.x] +
              parts[128 + threadIdx.x] + parts[192 + threadIdx.x];
    vmean[bh * DKH + threadIdx.x] = t * (1.0f / (float)LSEQ);
  }
}

// ---------------------------------------------------------------------------
// Full softmax(QK^T/8)V for the 27 selected queries of each (b,h)
// ---------------------------------------------------------------------------
__global__ __launch_bounds__(256)
void attn_top_kernel(const float* __restrict__ Q, const float* __restrict__ K,
                     const float* __restrict__ V, const int* __restrict__ topidx,
                     float* __restrict__ out_top) {
  __shared__ __align__(16) float qs[DKH];
  __shared__ float sc[LSEQ];
  __shared__ float red[256];
  const int tid = threadIdx.x;
  const int bh = blockIdx.x / NSAMP, u = blockIdx.x % NSAMP;
  const int b = bh >> 3, h = bh & 7;
  const int lq = topidx[bh * NSAMP + u];
  if (tid < DKH) qs[tid] = Q[(size_t)(b * LSEQ + lq) * DM + h * DKH + tid];
  __syncthreads();

  const float scale = 0.125f;
  float lmax = -3.4e38f;
  const float4* q4 = (const float4*)qs;
  for (int k = tid; k < LSEQ; k += 256) {
    const float4* k4 = (const float4*)(K + (size_t)(b * LSEQ + k) * DM + h * DKH);
    float d = 0.f;
#pragma unroll
    for (int i = 0; i < 16; ++i) {
      float4 a = q4[i], c = k4[i];
      d += a.x * c.x + a.y * c.y + a.z * c.z + a.w * c.w;
    }
    float v = d * scale;
    sc[k] = v;
    lmax = fmaxf(lmax, v);
  }
  red[tid] = lmax; __syncthreads();
  for (int off = 128; off; off >>= 1) {
    if (tid < off) red[tid] = fmaxf(red[tid], red[tid + off]);
    __syncthreads();
  }
  const float gmax = red[0];
  __syncthreads();

  float lsum = 0.f;
  for (int k = tid; k < LSEQ; k += 256) {
    float e = __expf(sc[k] - gmax);
    sc[k] = e;
    lsum += e;
  }
  red[tid] = lsum; __syncthreads();
  for (int off = 128; off; off >>= 1) {
    if (tid < off) red[tid] += red[tid + off];
    __syncthreads();
  }
  const float ginv = 1.0f / red[0];
  __syncthreads();

  const int d = tid & 63, c = tid >> 6;
  float acc = 0.f;
  for (int k = c; k < LSEQ; k += 4)
    acc += sc[k] * V[(size_t)(b * LSEQ + k) * DM + h * DKH + d];
  red[tid] = acc; __syncthreads();
  if (tid < 64) {
    float o = (red[tid] + red[tid + 64] + red[tid + 128] + red[tid + 192]) * ginv;
    out_top[(size_t)(bh * NSAMP + u) * DKH + tid] = o;
  }
}

// ---------------------------------------------------------------------------
// context(bf16) = broadcast(vmean); then scatter out_top rows at top_idx
// ---------------------------------------------------------------------------
__global__ void ctx_fill_kernel(const float* __restrict__ vmean,
                                bf16_t* __restrict__ ctx) {
  int g = blockIdx.x * 256 + threadIdx.x;    // 8 elems/thread, exact grid
  int col = (g << 3) & (DM - 1);
  int row = g >> 6;
  int b = row >> 12;
  int h = col >> 6, d = col & 63;
  const float* vm = vmean + (b * NH + h) * DKH + d;
  bf16_t tmp[8];
#pragma unroll
  for (int j = 0; j < 8; ++j) tmp[j] = f2bf(vm[j]);
  *(uint4*)(ctx + ((size_t)g << 3)) = *(uint4*)tmp;
}

__global__ void ctx_scatter_kernel(const float* __restrict__ out_top,
                                   const int* __restrict__ topidx,
                                   bf16_t* __restrict__ ctx) {
  int i = blockIdx.x * 256 + threadIdx.x;
  if (i >= BATCH * NH * NSAMP * DKH) return;
  int d = i & 63;
  int idx = i >> 6;
  int u = idx % NSAMP, bh = idx / NSAMP;
  int b = bh >> 3, h = bh & 7;
  int l = topidx[bh * NSAMP + u];
  ctx[(size_t)(b * LSEQ + l) * DM + h * DKH + d] = f2bf(out_top[i]);
}

// ---------------------------------------------------------------------------
// Host-side orchestration
// ---------------------------------------------------------------------------
static void run_gemm(const bf16_t* A, const bf16_t* Wt, const float* bias,
                     float* out1, bf16_t* out1bf, float* out2, const float* xs,
                     int row_off, hipStream_t s) {
  dim3 grid(DM / BN, (BATCH * LSEQ) / BM);
  gemm_bf16_kernel<<<grid, 256, 0, s>>>(A, Wt, bias, out1, out1bf, out2, xs, row_off);
}

static void run_attention(unsigned seed, float* Q, float* Kb, float* Vb,
                          bf16_t* CTXB, float* Mv, float* OT, float* VM,
                          int* TI, int* SI, hipStream_t s) {
  const int total = LSEQ * NSAMP;
  threefry_idx_kernel<<<(total / 2 + 255) / 256, 256, 0, s>>>(SI, 0u, seed, total);
  sample_score_kernel<<<(BATCH * NH * LSEQ) / 256, 256, 0, s>>>(Q, Kb, SI, Mv);
  topk_kernel<<<BATCH * NH, 256, 0, s>>>(Mv, TI);
  vmean_kernel<<<BATCH * NH, 256, 0, s>>>(Vb, VM);
  attn_top_kernel<<<BATCH * NH * NSAMP, 256, 0, s>>>(Q, Kb, Vb, TI, OT);
  ctx_fill_kernel<<<(BATCH * LSEQ * DM / 8) / 256, 256, 0, s>>>(VM, CTXB);
  ctx_scatter_kernel<<<(BATCH * NH * NSAMP * DKH + 255) / 256, 256, 0, s>>>(OT, TI, CTXB);
}

extern "C" void kernel_launch(void* const* d_in, const int* in_sizes, int n_in,
                              void* d_out, int out_size, void* d_ws, size_t ws_size,
                              hipStream_t stream) {
  (void)in_sizes; (void)n_in; (void)out_size; (void)ws_size;
  const float* xs = (const float*)d_in[0];
  const float* xd = (const float*)d_in[1];
  const float* xp = (const float*)d_in[2];
  const float* Wf[8] = {(const float*)d_in[3],  (const float*)d_in[4],
                        (const float*)d_in[5],  (const float*)d_in[6],
                        (const float*)d_in[11], (const float*)d_in[12],
                        (const float*)d_in[13], (const float*)d_in[14]};
  const float* Bf[8] = {(const float*)d_in[7],  (const float*)d_in[8],
                        (const float*)d_in[9],  (const float*)d_in[10],
                        (const float*)d_in[15], (const float*)d_in[16],
                        (const float*)d_in[17], (const float*)d_in[18]};

  char* wsb = (char*)d_ws;
  float*  Q    = (float*)(wsb + 0);
  float*  Kb   = (float*)(wsb + 33554432);
  float*  Vb   = (float*)(wsb + 67108864);
  bf16_t* CTXB = (bf16_t*)(wsb + 100663296);
  bf16_t* XP2B = (bf16_t*)(wsb + 117440512);
  bf16_t* AXP  = (bf16_t*)(wsb + 134217728);
  bf16_t* AXD  = (bf16_t*)(wsb + 150994944);
  bf16_t* WTS  = (bf16_t*)(wsb + 167772160);   // 8 slots of 512*512
  float*  Mv   = (float*)(wsb + 171966464);
  float*  OT   = (float*)(wsb + 172490752);
  float*  VM   = (float*)(wsb + 172711936);
  int*    TI   = (int*)  (wsb + 172720128);
  int*    SI   = (int*)  (wsb + 172724224);

  float* out = (float*)d_out;

  // ---- pre-convert activations + weights to bf16 (once) -------------------
  const int cvt_blocks = (BATCH * LSEQ * DM / 8) / 256;   // 4096
  cvt_bf16_kernel<<<cvt_blocks, 256, 0, stream>>>(xp, AXP);
  cvt_bf16_kernel<<<cvt_blocks, 256, 0, stream>>>(xd, AXD);
  for (int w = 0; w < 8; ++w)
    wt_convert_kernel<<<dim3(16, 16), 256, 0, stream>>>(Wf[w], WTS + (size_t)w * DM * DM);

  // ---- layer 0: q from xp, k/v from xd; xp2 -> XP2B + out rows [4096,8192)
  run_gemm(AXP, WTS + 0 * DM * DM, Bf[0], Q,  nullptr, nullptr, nullptr, 0, stream);
  run_gemm(AXD, WTS + 1 * DM * DM, Bf[1], Kb, nullptr, nullptr, nullptr, 0, stream);
  run_gemm(AXD, WTS + 2 * DM * DM, Bf[2], Vb, nullptr, nullptr, nullptr, 0, stream);
  run_attention(42u, Q, Kb, Vb, CTXB, Mv, OT, VM, TI, SI, stream);
  run_gemm(CTXB, WTS + 3 * DM * DM, Bf[3], nullptr, XP2B, out, xs, LSEQ, stream);

  // ---- layer 1: q from xd, k/v from xp2; xd2 -> out rows [0,4096) ----------
  run_gemm(AXD,  WTS + 4 * DM * DM, Bf[4], Q,  nullptr, nullptr, nullptr, 0, stream);
  run_gemm(XP2B, WTS + 5 * DM * DM, Bf[5], Kb, nullptr, nullptr, nullptr, 0, stream);
  run_gemm(XP2B, WTS + 6 * DM * DM, Bf[6], Vb, nullptr, nullptr, nullptr, 0, stream);
  run_attention(43u, Q, Kb, Vb, CTXB, Mv, OT, VM, TI, SI, stream);
  run_gemm(CTXB, WTS + 7 * DM * DM, Bf[7], nullptr, nullptr, out, xs, 0, stream);
}
